// Block_70643622084884
// MI455X (gfx1250) — compile-verified, matrix operand order checked by reading, not run on hardware
//
#include <hip/hip_runtime.h>
#include <hip/hip_bf16.h>

// ---------------------------------------------------------------------------
// GPT-2 style transformer block on gfx1250 (wave32, WMMA f16 -> f32 accum).
//   x -> LN1 -> QKV GEMM -> flash attention -> proj GEMM (+x) -> LN2
//     -> MLP1 GEMM (+GELU) -> MLP2 GEMM (+residual) -> out
// Global->LDS staging uses gfx1250 async-to-LDS DMA with double buffering.
// ---------------------------------------------------------------------------

#define D      768
#define NH     12
#define HD     64
#define NMLP   3072
#define TSEQ   4096
#define BATCH  2
#define ROWS   (BATCH * TSEQ)   // 8192
#define QKVW   (3 * D)          // 2304

typedef __attribute__((ext_vector_type(16))) _Float16 v16h;
typedef __attribute__((ext_vector_type(8)))  float    v8f;

union FragU { v16h v; uint4 u[2]; _Float16 e[16]; };

// gfx1250 async global->LDS copy of 16 bytes per lane (ASYNCcnt-tracked).
// ldsOff = wave-relative LDS byte offset (low 32 bits of flat shared addr).
__device__ __forceinline__ void async_copy16(unsigned ldsOff, const void* gsrc) {
  asm volatile("global_load_async_to_lds_b128 %0, %1, off"
               :: "v"(ldsOff), "v"(gsrc) : "memory");
}
__device__ __forceinline__ void async_wait() {
  asm volatile("s_wait_asynccnt 0x0" ::: "memory");
}
__device__ __forceinline__ unsigned lds_off(const void* p) {
  return (unsigned)(size_t)p;
}

// A/B fragment load, row-major source (LDS or global), 16-bit 16x32 layout:
// lanes 0-15: row = lane,    K = {k0..k0+7, k0+16..k0+23}
// lanes 16-31: row = lane-16, K = {k0+8..k0+15, k0+24..k0+31}
__device__ __forceinline__ v16h ldfrag_rm(const _Float16* base, int row0, int k0, int ld) {
  const int lane = threadIdx.x & 31;
  const _Float16* p = base + (size_t)(row0 + (lane & 15)) * ld + (k0 + ((lane >> 4) << 3));
  FragU u;
  u.u[0] = *(const uint4*)p;
  u.u[1] = *(const uint4*)(p + 16);
  return u.v;
}

// B fragment where the source tile is row-major [K][ld] and each lane owns
// column n0 + (lane&15) (used for P@V where V sits row-major in LDS).
__device__ __forceinline__ v16h ldfrag_col(const _Float16* sm, int n0, int ld) {
  const int lane = threadIdx.x & 31;
  const int n  = n0 + (lane & 15);
  const int kb = (lane >> 4) << 3;
  FragU u;
#pragma unroll
  for (int e = 0; e < 8; ++e) u.e[e]     = sm[(kb + e) * ld + n];
#pragma unroll
  for (int e = 0; e < 8; ++e) u.e[8 + e] = sm[(16 + kb + e) * ld + n];
  return u.v;
}

__device__ __forceinline__ v8f wmma_f16(v16h a, v16h b, v8f c) {
  return __builtin_amdgcn_wmma_f32_16x16x32_f16(false, a, false, b, (short)0, c, false, false);
}

// ---------------------------------------------------------------------------
// fp32 -> fp16 elementwise convert (weights)
// ---------------------------------------------------------------------------
__global__ void f32_to_f16_kernel(const float* __restrict__ in,
                                  _Float16* __restrict__ out, int n) {
  int i = blockIdx.x * blockDim.x + threadIdx.x;
  if (i < n) out[i] = (_Float16)in[i];
}

// ---------------------------------------------------------------------------
// LayerNorm: one wave per row of 768, output f16
// ---------------------------------------------------------------------------
__global__ __launch_bounds__(256) void ln_f16_kernel(const float* __restrict__ x,
                                                     const float* __restrict__ g,
                                                     const float* __restrict__ be,
                                                     _Float16* __restrict__ out) {
  const int row  = blockIdx.x * 8 + (threadIdx.x >> 5);
  const int lane = threadIdx.x & 31;
  const float* xr = x + (size_t)row * D;
  float v[24];
  float s = 0.f;
#pragma unroll
  for (int i = 0; i < 24; ++i) { v[i] = xr[lane + i * 32]; s += v[i]; }
#pragma unroll
  for (int mask = 16; mask >= 1; mask >>= 1) s += __shfl_xor(s, mask, 32);
  const float mu = s * (1.0f / D);
  float s2 = 0.f;
#pragma unroll
  for (int i = 0; i < 24; ++i) { float d = v[i] - mu; s2 += d * d; }
#pragma unroll
  for (int mask = 16; mask >= 1; mask >>= 1) s2 += __shfl_xor(s2, mask, 32);
  const float rstd = rsqrtf(s2 * (1.0f / D) + 1e-6f);
  _Float16* orow = out + (size_t)row * D;
#pragma unroll
  for (int i = 0; i < 24; ++i) {
    int c = lane + i * 32;
    orow[c] = (_Float16)((v[i] - mu) * rstd * g[c] + be[c]);
  }
}

// ---------------------------------------------------------------------------
// Tiled WMMA GEMM: C[M,N] = A[M,K] @ W[N,K]^T + bias, epilogue variants
//   EPI=0: f16 store (qkv)      EPI=1: f32 store + residual (proj / out)
//   EPI=2: exact GELU, f16 store (mlp hidden)
// Block: 256 thr = 8 waves; tile 128x128x32; wave = 32x64 (2x4 WMMA tiles).
// Double-buffered LDS staged by async-to-LDS DMA.
// ---------------------------------------------------------------------------
template<int EPI>
__global__ __launch_bounds__(256) void gemm_wmma_kernel(
    const _Float16* __restrict__ A, const _Float16* __restrict__ W,
    const float* __restrict__ bias, const float* __restrict__ resid,
    float* __restrict__ outF, _Float16* __restrict__ outH,
    int M, int N, int K) {
  __shared__ _Float16 sA[2][128 * 40];
  __shared__ _Float16 sB[2][128 * 40];
  const int tid  = threadIdx.x;
  const int wave = tid >> 5, lane = tid & 31;
  const int wm = (wave & 3) * 32;   // 4 waves along M
  const int wn = (wave >> 2) * 64;  // 2 waves along N
  const int rowBase = blockIdx.y * 128, colBase = blockIdx.x * 128;

  v8f zero = {0.f, 0.f, 0.f, 0.f, 0.f, 0.f, 0.f, 0.f};
  v8f acc[2][4];
#pragma unroll
  for (int i = 0; i < 2; ++i)
#pragma unroll
    for (int j = 0; j < 4; ++j) acc[i][j] = zero;

  auto stage = [&](int buf, int kb) {
#pragma unroll
    for (int i = tid; i < 512; i += 256) {          // 128 rows x 32 halves / tile
      int r = i >> 2, c = (i & 3) << 3;
      async_copy16(lds_off(&sA[buf][r * 40 + c]),
                   &A[(size_t)(rowBase + r) * K + kb + c]);
      async_copy16(lds_off(&sB[buf][r * 40 + c]),
                   &W[(size_t)(colBase + r) * K + kb + c]);
    }
  };

  stage(0, 0);
  async_wait();
  __syncthreads();

  const int nk = K / 32;
  for (int it = 0; it < nk; ++it) {
    const int cur = it & 1;
    if (it + 1 < nk) stage(cur ^ 1, (it + 1) * 32);

    const _Float16* cA = sA[cur];
    const _Float16* cB = sB[cur];
    v16h af[2], bf[4];
    af[0] = ldfrag_rm(cA, wm,      0, 40);
    af[1] = ldfrag_rm(cA, wm + 16, 0, 40);
#pragma unroll
    for (int nt = 0; nt < 4; ++nt) bf[nt] = ldfrag_rm(cB, wn + nt * 16, 0, 40);
#pragma unroll
    for (int mt = 0; mt < 2; ++mt)
#pragma unroll
      for (int nt = 0; nt < 4; ++nt)
        acc[mt][nt] = wmma_f16(af[mt], bf[nt], acc[mt][nt]);

    async_wait();       // own wave's prefetch landed
    __syncthreads();    // everyone done reading cur + prefetch visible
  }

  const int cN = lane & 15, rOff = (lane >> 4) * 8;
#pragma unroll
  for (int mt = 0; mt < 2; ++mt)
#pragma unroll
    for (int nt = 0; nt < 4; ++nt)
#pragma unroll
      for (int r = 0; r < 8; ++r) {
        int row = rowBase + wm + mt * 16 + rOff + r;
        int col = colBase + wn + nt * 16 + cN;
        float v = acc[mt][nt][r] + bias[col];
        if (EPI == 0) {
          outH[(size_t)row * N + col] = (_Float16)v;
        } else if (EPI == 1) {
          outF[(size_t)row * N + col] = v + resid[(size_t)row * N + col];
        } else {
          v = 0.5f * v * (1.0f + erff(v * 0.70710678118f));
          outH[(size_t)row * N + col] = (_Float16)v;
        }
      }
}

// ---------------------------------------------------------------------------
// Flash attention (non-causal): grid (T/64, NH, B), block 128 = 4 waves.
// Each wave owns 16 Q rows x full HD=64; streams 32-key tiles.
// K/V double-buffered via async-to-LDS DMA.
// ---------------------------------------------------------------------------
__global__ __launch_bounds__(128) void flash_attn_kernel(
    const _Float16* __restrict__ qkv, _Float16* __restrict__ attn) {
  __shared__ _Float16 sK[2][32 * 72];
  __shared__ _Float16 sV[2][32 * 72];
  __shared__ _Float16 sP[4 * 16 * 40];
  const int tid  = threadIdx.x;
  const int wave = tid >> 5, lane = tid & 31;
  const int b = blockIdx.z, hh = blockIdx.y;
  const int q0 = blockIdx.x * 64 + wave * 16;
  const size_t bBase = (size_t)b * TSEQ * QKVW;
  const _Float16* Qp = qkv + bBase + hh * HD;
  const _Float16* Kp = Qp + D;
  const _Float16* Vp = Qp + 2 * D;

  const v16h aq0 = ldfrag_rm(Qp, q0, 0,  QKVW);
  const v16h aq1 = ldfrag_rm(Qp, q0, 32, QKVW);

  v8f zero = {0.f, 0.f, 0.f, 0.f, 0.f, 0.f, 0.f, 0.f};
  v8f acc[4];
#pragma unroll
  for (int vt = 0; vt < 4; ++vt) acc[vt] = zero;
  float m[8], l[8];
#pragma unroll
  for (int r = 0; r < 8; ++r) { m[r] = -1e30f; l[r] = 0.f; }

  _Float16* myP = sP + wave * 16 * 40;
  const int cN = lane & 15, rOff = (lane >> 4) * 8;

  auto stageKV = [&](int buf, int j) {
#pragma unroll
    for (int i = tid; i < 256; i += 128) {          // 32 rows x 64 halves each
      int rr = i >> 3, c = (i & 7) << 3;
      async_copy16(lds_off(&sK[buf][rr * 72 + c]),
                   &Kp[(size_t)(j + rr) * QKVW + c]);
      async_copy16(lds_off(&sV[buf][rr * 72 + c]),
                   &Vp[(size_t)(j + rr) * QKVW + c]);
    }
  };

  stageKV(0, 0);
  async_wait();
  __syncthreads();

  const int nit = TSEQ / 32;
  for (int it = 0; it < nit; ++it) {
    const int cur = it & 1;
    if (it + 1 < nit) stageKV(cur ^ 1, (it + 1) * 32);
    const _Float16* cK = sK[cur];
    const _Float16* cV = sV[cur];

    // S = Q @ K^T : hoist all four K fragments, then the WMMA chain
    v16h bk00 = ldfrag_rm(cK, 0,  0,  72);
    v16h bk01 = ldfrag_rm(cK, 0,  32, 72);
    v16h bk10 = ldfrag_rm(cK, 16, 0,  72);
    v16h bk11 = ldfrag_rm(cK, 16, 32, 72);
    v8f s[2];
    s[0] = wmma_f16(aq0, bk00, zero);
    s[1] = wmma_f16(aq0, bk10, zero);
    s[0] = wmma_f16(aq1, bk01, s[0]);
    s[1] = wmma_f16(aq1, bk11, s[1]);

    // online softmax (row stats via width-16 xor reductions)
#pragma unroll
    for (int r = 0; r < 8; ++r) {
      float s0 = s[0][r] * 0.125f;
      float s1 = s[1][r] * 0.125f;
      float mx = fmaxf(s0, s1);
#pragma unroll
      for (int mask = 8; mask >= 1; mask >>= 1) mx = fmaxf(mx, __shfl_xor(mx, mask, 32));
      float mnew  = fmaxf(m[r], mx);
      float alpha = __expf(m[r] - mnew);
      m[r] = mnew;
      s0 = __expf(s0 - mnew);
      s1 = __expf(s1 - mnew);
      float ps = s0 + s1;
#pragma unroll
      for (int mask = 8; mask >= 1; mask >>= 1) ps += __shfl_xor(ps, mask, 32);
      l[r] = l[r] * alpha + ps;
#pragma unroll
      for (int vt = 0; vt < 4; ++vt) acc[vt][r] *= alpha;
      myP[(rOff + r) * 40 + cN]      = (_Float16)s0;
      myP[(rOff + r) * 40 + 16 + cN] = (_Float16)s1;
    }
    __syncthreads();   // P visible (and sK reads done)

    // acc += P @ V : hoist P and all four V fragments
    const v16h ap = ldfrag_rm(myP, 0, 0, 40);
    v16h bv[4];
#pragma unroll
    for (int vt = 0; vt < 4; ++vt) bv[vt] = ldfrag_col(cV, vt * 16, 72);
#pragma unroll
    for (int vt = 0; vt < 4; ++vt) acc[vt] = wmma_f16(ap, bv[vt], acc[vt]);

    async_wait();      // own wave's K/V prefetch landed
    __syncthreads();   // all waves done with cur buffers
  }

#pragma unroll
  for (int vt = 0; vt < 4; ++vt)
#pragma unroll
    for (int r = 0; r < 8; ++r) {
      int row = q0 + rOff + r;
      int col = hh * HD + vt * 16 + cN;
      attn[((size_t)b * TSEQ + row) * D + col] = (_Float16)(acc[vt][r] / l[r]);
    }
}

// ---------------------------------------------------------------------------
// Launch
// ---------------------------------------------------------------------------
extern "C" void kernel_launch(void* const* d_in, const int* in_sizes, int n_in,
                              void* d_out, int out_size, void* d_ws, size_t ws_size,
                              hipStream_t stream) {
  const float* x      = (const float*)d_in[0];
  const float* W_qkv  = (const float*)d_in[1];
  const float* b_qkv  = (const float*)d_in[2];
  const float* W_proj = (const float*)d_in[3];
  const float* b_proj = (const float*)d_in[4];
  const float* W1     = (const float*)d_in[5];
  const float* b1     = (const float*)d_in[6];
  const float* W2     = (const float*)d_in[7];
  const float* b2     = (const float*)d_in[8];
  const float* g1     = (const float*)d_in[9];
  const float* beta1  = (const float*)d_in[10];
  const float* g2     = (const float*)d_in[11];
  const float* beta2  = (const float*)d_in[12];
  float* out = (float*)d_out;

  char* p = (char*)d_ws;
  auto alloc = [&](size_t bytes) -> void* {
    void* r = (void*)p;
    p += (bytes + 255) & ~(size_t)255;
    return r;
  };
  _Float16* wq  = (_Float16*)alloc((size_t)QKVW * D * 2);
  _Float16* wp  = (_Float16*)alloc((size_t)D * D * 2);
  _Float16* w1h = (_Float16*)alloc((size_t)NMLP * D * 2);
  _Float16* w2h = (_Float16*)alloc((size_t)D * NMLP * 2);
  _Float16* h1  = (_Float16*)alloc((size_t)ROWS * D * 2);
  _Float16* qkv = (_Float16*)alloc((size_t)ROWS * QKVW * 2);
  _Float16* att = (_Float16*)alloc((size_t)ROWS * D * 2);
  float*    x1  = (float*)   alloc((size_t)ROWS * D * 4);
  _Float16* h2  = (_Float16*)alloc((size_t)ROWS * D * 2);
  _Float16* mid = (_Float16*)alloc((size_t)ROWS * NMLP * 2);

  auto cvt = [&](const float* src, _Float16* dst, int n) {
    f32_to_f16_kernel<<<(n + 255) / 256, 256, 0, stream>>>(src, dst, n);
  };
  cvt(W_qkv,  wq,  QKVW * D);
  cvt(W_proj, wp,  D * D);
  cvt(W1,     w1h, NMLP * D);
  cvt(W2,     w2h, D * NMLP);

  // LN1
  ln_f16_kernel<<<ROWS / 8, 256, 0, stream>>>(x, g1, beta1, h1);
  // QKV GEMM (f16 out)
  gemm_wmma_kernel<0><<<dim3(QKVW / 128, ROWS / 128), 256, 0, stream>>>(
      h1, wq, b_qkv, nullptr, nullptr, qkv, ROWS, QKVW, D);
  // attention
  flash_attn_kernel<<<dim3(TSEQ / 64, NH, BATCH), 128, 0, stream>>>(qkv, att);
  // proj GEMM + residual (f32 out -> x1)
  gemm_wmma_kernel<1><<<dim3(D / 128, ROWS / 128), 256, 0, stream>>>(
      att, wp, b_proj, x, x1, nullptr, ROWS, D, D);
  // LN2
  ln_f16_kernel<<<ROWS / 8, 256, 0, stream>>>(x1, g2, beta2, h2);
  // MLP1 GEMM + GELU (f16 out)
  gemm_wmma_kernel<2><<<dim3(NMLP / 128, ROWS / 128), 256, 0, stream>>>(
      h2, w1h, b1, nullptr, nullptr, mid, ROWS, NMLP, D);
  // MLP2 GEMM + residual -> final output
  gemm_wmma_kernel<1><<<dim3(D / 128, ROWS / 128), 256, 0, stream>>>(
      mid, w2h, b2, x1, out, nullptr, ROWS, D, NMLP);
}
